// MultiHeadSelfAttentionWithRPE_78219944395273
// MI455X (gfx1250) — matrix operand order repaired and checked by reading, not compiled, not run on hardware
//
#include <hip/hip_runtime.h>
#include <hip/hip_bf16.h>

typedef __attribute__((ext_vector_type(16))) _Float16     v16h;
typedef __attribute__((ext_vector_type(8)))  float        v8f;
typedef __attribute__((ext_vector_type(4)))  unsigned int v4u;
typedef __attribute__((ext_vector_type(8)))  int          v8i;
typedef __attribute__((ext_vector_type(4)))  int          v4i;

#define E_  768
#define H_  12
#define D_  64
#define N_  8
#define L_  512
#define P_  1023          // 2*MAXLEN-1 distinct relative positions
#define NL_ (N_ * L_)     // 4096

static __device__ __forceinline__ v8f zero8() { v8f z = {}; return z; }

static __device__ __forceinline__ v8f wmma_f16(v16h a, v16h b, v8f c) {
  return __builtin_amdgcn_wmma_f32_16x16x32_f16(false, a, false, b, (short)0, c,
                                                false, false);
}

// ---------------------------------------------------------------------------
// Tensor Data Mover: async 2-D tile load Global -> LDS (CDNA5 ISA).
// Descriptor group 0: count=1 | lds_addr | global_addr[56:0] | type=2.
// Group 1: data_size, tensor_dim0/1, tile_dim0/1, tensor_dim0_stride.
// Groups 2/3 zero (<=2-D tile). Tracked with TENSORcnt.
// dsize_code: 0=1B 1=2B 2=4B 3=8B. Dims/strides in data_size units.
// ---------------------------------------------------------------------------
static __device__ __forceinline__ void tdm_load_2d(
    unsigned lds_off, const void* gaddr, unsigned dsize_code,
    unsigned tile_d0, unsigned tile_d1, unsigned long long stride_d0,
    unsigned tens_d0, unsigned tens_d1) {
  unsigned long long ga = (unsigned long long)(uintptr_t)gaddr;
  v4u g0;
  g0[0] = 1u;                                             // count=1, user D#
  g0[1] = lds_off;                                        // LDS byte address
  g0[2] = (unsigned)ga;                                   // global addr [31:0]
  g0[3] = (unsigned)((ga >> 32) & 0x1FFFFFFull) | (2u << 30);  // [56:32]|type=2
  v8i g1;
  g1[0] = (int)(dsize_code << 16);                        // data_size
  g1[1] = (int)((tens_d0 & 0xFFFFu) << 16);               // tensor_dim0 lo16
  g1[2] = (int)((tens_d0 >> 16) | ((tens_d1 & 0xFFFFu) << 16));
  g1[3] = (int)((tens_d1 >> 16) | (tile_d0 << 16));       // tile_dim0
  g1[4] = (int)(tile_d1 & 0xFFFFu);                       // tile_dim1 (dim2=0)
  g1[5] = (int)(unsigned)(stride_d0 & 0xFFFFFFFFull);     // dim0_stride lo32
  g1[6] = (int)(unsigned)((stride_d0 >> 32) & 0xFFFFull); // dim0_stride hi16
  g1[7] = 0;
  v4i gz = {};
#if __clang_major__ >= 23
  v8i gz8 = {};
  __builtin_amdgcn_tensor_load_to_lds(g0, g1, gz, gz, gz8, 0);
#else
  __builtin_amdgcn_tensor_load_to_lds(g0, g1, gz, gz, 0);
#endif
}

// ---------------------------------------------------------------------------
// Sinusoidal table: pe[p, 2i] = sin(p*w_i), pe[p, 2i+1] = cos(p*w_i)
// ---------------------------------------------------------------------------
__global__ void build_pe_kernel(float* __restrict__ pe) {
  int i = blockIdx.x * blockDim.x + threadIdx.x;
  const int total = P_ * (E_ / 2);
  if (i >= total) return;
  int p  = i / (E_ / 2);
  int pr = i - p * (E_ / 2);
  float w   = __expf((float)(2 * pr) * (-9.210340371976184f / (float)E_));
  float ang = (float)p * w;
  pe[p * E_ + 2 * pr]     = __sinf(ang);
  pe[p * E_ + 2 * pr + 1] = __cosf(ang);
}

// ---------------------------------------------------------------------------
// Generic WMMA GEMM:  C(M x 768) = A(M x 768, f32) * W(768 x 768, f32) + bias
// Block: 256 threads = 8 waves; tile 128(M) x 64(N); each wave 16 x 64.
// W tile (32x64 f32) DMA'd by the TDM into double-buffered LDS one K-step
// ahead (s_wait_tensorcnt pipelining); VALU converts to f16 fragment order.
// MODE 0: f16 out0 = acc + bias0
// MODE 1: f16 out0 = acc + bias0 + bias1 ; f16 out1 = acc + bias0 + bias2
// MODE 2: f32 out32 = acc + bias0
// ---------------------------------------------------------------------------
template <int MODE>
__global__ __launch_bounds__(256) void gemm_wmma_kernel(
    const float* __restrict__ A, const float* __restrict__ W,
    const float* __restrict__ bias0, const float* __restrict__ bias1,
    const float* __restrict__ bias2,
    _Float16* __restrict__ out0, _Float16* __restrict__ out1,
    float* __restrict__ out32, int M) {
  __shared__ __align__(16) float    Wt[2][32 * 64];   // TDM landing buffers
  __shared__ __align__(32) _Float16 Bh[4 * 32 * 16];  // fragment-order f16

  const int tid  = threadIdx.x;
  const int wave = tid >> 5;
  const int lane = tid & 31;
  const int lh   = lane & 15;
  const int hi   = lane >> 4;
  const int m0   = blockIdx.x * 128 + wave * 16;
  const int c0   = blockIdx.y * 64;

  const unsigned wt_off0 = (unsigned)(uintptr_t)(void*)&Wt[0][0];
  const unsigned wt_off1 = (unsigned)(uintptr_t)(void*)&Wt[1][0];

  int arow = m0 + lh;
  if (arow > M - 1) arow = M - 1;            // clamp ragged M (stores bounded)
  const float* aptr = A + (size_t)arow * E_;

  v8f acc[4];
#pragma unroll
  for (int nt = 0; nt < 4; ++nt) acc[nt] = zero8();

  // Prime the pipeline: DMA first 32x64 W tile into buffer 0.
  if (wave == 0)
    tdm_load_2d(wt_off0, W + (size_t)0 * E_ + c0, 2, 64, 32, E_, 64, 32);

  for (int k0 = 0; k0 < E_; k0 += 32) {
    __syncthreads();   // prior iteration's Bh reads + Wt conversion complete
    if (wave == 0) {
      if (k0 + 32 < E_) {
        unsigned nxt = (((k0 >> 5) + 1) & 1) ? wt_off1 : wt_off0;
        tdm_load_2d(nxt, W + (size_t)(k0 + 32) * E_ + c0, 2, 64, 32, E_, 64, 32);
        __builtin_amdgcn_s_wait_tensorcnt(1);  // current tile landed
      } else {
        __builtin_amdgcn_s_wait_tensorcnt(0);
      }
    }
    __syncthreads();   // Wt[cur] visible to all waves

    // f32 LDS tile -> f16 fragment order:
    // element (lane, h): Wtile[((h&7) + (h>>3)*16 + (lane>>4)*8) * 64 + nt*16 + (lane&15)]
    const float* wt = Wt[(k0 >> 5) & 1];
#pragma unroll
    for (int u = 0; u < 8; ++u) {
      int f   = tid * 8 + u;                 // 0..2047
      int h   = f & 15;
      int ln  = (f >> 4) & 31;
      int nt  = f >> 9;
      int K   = (h & 7) + ((h >> 3) << 4) + ((ln >> 4) << 3);
      int col = (nt << 4) + (ln & 15);
      Bh[f]   = (_Float16)wt[K * 64 + col];
    }
    __syncthreads();

    // A fragment: 16-bit A 16x32 layout, K = (h&7) + (h>>3)*16 + hi*8
    const float* ap = aptr + k0 + hi * 8;
    v16h af;
#pragma unroll
    for (int t = 0; t < 8; ++t) af[t] = (_Float16)ap[t];
#pragma unroll
    for (int t = 0; t < 8; ++t) af[8 + t] = (_Float16)ap[16 + t];

    const v16h* bf = (const v16h*)Bh;
#pragma unroll
    for (int nt = 0; nt < 4; ++nt)
      acc[nt] = wmma_f16(af, bf[nt * 32 + lane], acc[nt]);
  }

#pragma unroll
  for (int nt = 0; nt < 4; ++nt) {
    int col = c0 + nt * 16 + lh;
#pragma unroll
    for (int r = 0; r < 8; ++r) {
      int row = m0 + hi * 8 + r;
      if (row < M) {
        float v = acc[nt][r];
        if (MODE == 0) {
          out0[(size_t)row * E_ + col] = (_Float16)(v + bias0[col]);
        } else if (MODE == 1) {
          out0[(size_t)row * E_ + col] = (_Float16)(v + bias0[col] + bias1[col]);
          out1[(size_t)row * E_ + col] = (_Float16)(v + bias0[col] + bias2[col]);
        } else {
          out32[(size_t)row * E_ + col] = v + bias0[col];
        }
      }
    }
  }
}

// ---------------------------------------------------------------------------
// Fused flash attention with relative-position bias (Transformer-XL style).
// One wave per (n, h, 16 query rows). BD computed once per q-tile as a
// 16 x 528 WMMA GEMM over relative positions p in [q0, q0+527], kept in LDS
// (p = q - k + 511 is never clipped for L=512). _rel_shift is identity.
// V tiles (32 keys x 64 dims, f16) are TDM'd into double-buffered LDS one
// iteration ahead; the wait is deferred until just before the PV WMMAs so
// the AC/softmax work hides the DMA latency.
// ---------------------------------------------------------------------------
__global__ __launch_bounds__(32) void flash_rpe_kernel(
    const _Float16* __restrict__ qw, const _Float16* __restrict__ qr,
    const _Float16* __restrict__ kf, const _Float16* __restrict__ vf,
    const _Float16* __restrict__ relp, float* __restrict__ ctx) {
  const int q0   = blockIdx.x * 16;
  const int h    = blockIdx.y;
  const int n    = blockIdx.z;
  const int lane = threadIdx.x & 31;
  const int lh   = lane & 15;
  const int hi   = lane >> 4;

  __shared__ float    bdbuf[16][544];     // 528 used, padded stride (~34KB)
  __shared__ _Float16 plds[16][32];       // prob tile for lane transpose
  __shared__ __align__(16) _Float16 Vt[2][32 * 64];  // TDM V-tile buffers

  const unsigned vt_off0 = (unsigned)(uintptr_t)(void*)&Vt[0][0];
  const unsigned vt_off1 = (unsigned)(uintptr_t)(void*)&Vt[1][0];

  // Q fragments (both bias variants), 2 K-chunks of 32
  v16h aqw[2], aqr[2];
  {
    const _Float16* qwp = qw + (size_t)(n * L_ + q0 + lh) * E_ + h * D_;
    const _Float16* qrp = qr + (size_t)(n * L_ + q0 + lh) * E_ + h * D_;
#pragma unroll
    for (int c = 0; c < 2; ++c) {
      int base = c * 32 + hi * 16;
#pragma unroll
      for (int t = 0; t < 8; ++t) {
        aqw[c][t]     = qwp[base + t];
        aqr[c][t]     = qrp[base + t];
        aqw[c][8 + t] = qwp[base + 8 + t];
        aqr[c][8 + t] = qrp[base + 8 + t];
      }
    }
  }

  // Prime V pipeline: DMA keys [0,32) x dims [h*64, h*64+64) into buffer 0.
  tdm_load_2d(vt_off0, vf + (size_t)(n * L_) * E_ + h * D_, 1, 64, 32, E_, 64, 32);

  // BD strip: 33 tiles x 16 relative positions -> LDS (overlaps first DMA)
#pragma unroll 1
  for (int t = 0; t < 33; ++t) {
    int p = q0 + t * 16 + lh;
    if (p > P_ - 1) p = P_ - 1;           // only padded last column
    const _Float16* rp = relp + (size_t)p * E_ + h * D_;
    v8f c = zero8();
    c = wmma_f16(aqr[0], *(const v16h*)(rp + hi * 16), c);
    c = wmma_f16(aqr[1], *(const v16h*)(rp + 32 + hi * 16), c);
#pragma unroll
    for (int r = 0; r < 8; ++r) bdbuf[hi * 8 + r][t * 16 + lh] = c[r];
  }

  v8f o[4], m, lsum;
#pragma unroll
  for (int r = 0; r < 8; ++r) { m[r] = -3.0e38f; lsum[r] = 0.0f; }
#pragma unroll
  for (int nt = 0; nt < 4; ++nt) o[nt] = zero8();

  for (int j0 = 0; j0 < L_; j0 += 32) {
    const int ib = (j0 >> 5) & 1;
    if (j0 + 32 < L_)   // kick off next V tile; also prefetch next K rows
      tdm_load_2d(ib ? vt_off0 : vt_off1,
                  vf + (size_t)(n * L_ + j0 + 32) * E_ + h * D_, 1, 64, 32,
                  E_, 64, 32);
    if (j0 + 32 < L_)
      __builtin_prefetch(kf + (size_t)(n * L_ + j0 + 32 + lh) * E_ + h * D_, 0, 0);

    v8f s[2];
#pragma unroll
    for (int tt = 0; tt < 2; ++tt) {
      int key = j0 + tt * 16 + lh;
      const _Float16* kp = kf + (size_t)(n * L_ + key) * E_ + h * D_;
      v8f c = zero8();
      c = wmma_f16(aqw[0], *(const v16h*)(kp + hi * 16), c);
      c = wmma_f16(aqw[1], *(const v16h*)(kp + 32 + hi * 16), c);
#pragma unroll
      for (int r = 0; r < 8; ++r) {
        int idx = (hi * 8 + r) - (tt * 16 + lh) - j0 + 511;  // in [0, 526]
        c[r] = (c[r] + bdbuf[hi * 8 + r][idx]) * 0.125f;     // / sqrt(64)
      }
      s[tt] = c;
    }

    // online softmax: row reductions across 16-lane halves
    v8f mt;
#pragma unroll
    for (int r = 0; r < 8; ++r) mt[r] = fmaxf(s[0][r], s[1][r]);
#pragma unroll
    for (int d = 1; d < 16; d <<= 1)
#pragma unroll
      for (int r = 0; r < 8; ++r) mt[r] = fmaxf(mt[r], __shfl_xor(mt[r], d, 32));

    v8f mnew, scale, p0, p1, rs;
#pragma unroll
    for (int r = 0; r < 8; ++r) {
      mnew[r]  = fmaxf(m[r], mt[r]);
      scale[r] = __expf(m[r] - mnew[r]);
      p0[r]    = __expf(s[0][r] - mnew[r]);
      p1[r]    = __expf(s[1][r] - mnew[r]);
      rs[r]    = p0[r] + p1[r];
    }
#pragma unroll
    for (int d = 1; d < 16; d <<= 1)
#pragma unroll
      for (int r = 0; r < 8; ++r) rs[r] += __shfl_xor(rs[r], d, 32);
#pragma unroll
    for (int r = 0; r < 8; ++r) {
      lsum[r] = lsum[r] * scale[r] + rs[r];
      m[r]    = mnew[r];
    }
#pragma unroll
    for (int nt = 0; nt < 4; ++nt)
#pragma unroll
      for (int r = 0; r < 8; ++r) o[nt][r] *= scale[r];

    // C-layout -> A-fragment transpose through LDS (same-wave DS in order)
#pragma unroll
    for (int r = 0; r < 8; ++r) {
      plds[hi * 8 + r][lh]      = (_Float16)p0[r];
      plds[hi * 8 + r][16 + lh] = (_Float16)p1[r];
    }
    v16h pa;
#pragma unroll
    for (int t = 0; t < 16; ++t) {
      int kk = t + hi * 16;
      pa[t]  = plds[lh][kk];
    }

    // Current V tile must have landed (next tile may still be in flight).
    if (j0 + 32 < L_)
      __builtin_amdgcn_s_wait_tensorcnt(1);
    else
      __builtin_amdgcn_s_wait_tensorcnt(0);

    // PV: probs(16x32) x V(32x64), V fragments gathered from LDS
#pragma unroll
    for (int nt = 0; nt < 4; ++nt) {
      v16h vb;  // B frag: K = key = hi*16 + t ; N = d = nt*16 + lh
#pragma unroll
      for (int t = 0; t < 16; ++t)
        vb[t] = Vt[ib][(hi * 16 + t) * 64 + nt * 16 + lh];
      o[nt] = wmma_f16(pa, vb, o[nt]);
    }
  }

#pragma unroll
  for (int nt = 0; nt < 4; ++nt)
#pragma unroll
    for (int r = 0; r < 8; ++r) {
      int row = q0 + hi * 8 + r;
      ctx[(size_t)(n * L_ + row) * E_ + h * D_ + nt * 16 + lh] =
          o[nt][r] / lsum[r];
    }
}

// ---------------------------------------------------------------------------
extern "C" void kernel_launch(void* const* d_in, const int* in_sizes, int n_in,
                              void* d_out, int out_size, void* d_ws,
                              size_t ws_size, hipStream_t stream) {
  const float* values = (const float*)d_in[0];
  const float* keys   = (const float*)d_in[1];
  const float* query  = (const float*)d_in[2];
  const float* Wq     = (const float*)d_in[3];
  const float* bq     = (const float*)d_in[4];
  const float* Wk     = (const float*)d_in[5];
  const float* bk     = (const float*)d_in[6];
  const float* Wv     = (const float*)d_in[7];
  const float* bv     = (const float*)d_in[8];
  const float* Wo     = (const float*)d_in[9];
  const float* bo     = (const float*)d_in[10];
  const float* Wpos   = (const float*)d_in[11];
  const float* bpos   = (const float*)d_in[12];
  const float* rw     = (const float*)d_in[13];
  const float* rr     = (const float*)d_in[14];
  float* out          = (float*)d_out;

  char* ws = (char*)d_ws;
  _Float16* qwv  = (_Float16*)(ws + 0);          // 4096x768 f16 (q + r_w_bias)
  _Float16* qrv  = (_Float16*)(ws + 6291456);    // 4096x768 f16 (q + r_r_bias)
  _Float16* kf   = (_Float16*)(ws + 12582912);   // 4096x768 f16
  _Float16* vf   = (_Float16*)(ws + 18874368);   // 4096x768 f16
  _Float16* relp = (_Float16*)(ws + 25165824);   // 1023x768 f16
  float*    pe   = (float*)(ws + 26737152);      // 1023x768 f32
  float*    ctx  = (float*)(ws + 29879808);      // 4096x768 f32 (end ~42.5MB)

  build_pe_kernel<<<(P_ * (E_ / 2) + 255) / 256, 256, 0, stream>>>(pe);

  dim3 gM((NL_ + 127) / 128, E_ / 64);   // (32, 12)
  dim3 gP((P_ + 127) / 128, E_ / 64);    // (8, 12)

  gemm_wmma_kernel<1><<<gM, 256, 0, stream>>>(query, Wq, bq, rw, rr,
                                              qwv, qrv, nullptr, NL_);
  gemm_wmma_kernel<0><<<gM, 256, 0, stream>>>(keys, Wk, bk, nullptr, nullptr,
                                              kf, nullptr, nullptr, NL_);
  gemm_wmma_kernel<0><<<gM, 256, 0, stream>>>(values, Wv, bv, nullptr, nullptr,
                                              vf, nullptr, nullptr, NL_);
  gemm_wmma_kernel<0><<<gP, 256, 0, stream>>>(pe, Wpos, bpos, nullptr, nullptr,
                                              relp, nullptr, nullptr, P_);

  dim3 gF(L_ / 16, H_, N_);              // (32, 12, 8)
  flash_rpe_kernel<<<gF, 32, 0, stream>>>(qwv, qrv, kf, vf, relp, ctx);

  gemm_wmma_kernel<2><<<gM, 256, 0, stream>>>(ctx, Wo, bo, nullptr, nullptr,
                                              nullptr, nullptr, out, NL_);
}
